// MultiTaskGNN_146028888368
// MI455X (gfx1250) — matrix operand order repaired and checked by reading, not compiled, not run on hardware
//
#include <hip/hip_runtime.h>
#include <hip/hip_bf16.h>

// ---------------------------------------------------------------------------
// MultiTaskGNN on MI455X (gfx1250).
// Main GEMMs (X[100k,256] @ W[256,256]) run on WMMA bf16->f32 matrix cores.
// Everything else (scatter, BN reductions, pooling, small MLPs) is
// bandwidth/atomic bound and runs as plain fp32 kernels.
// ---------------------------------------------------------------------------

typedef __attribute__((ext_vector_type(16))) __bf16 v16bf;
typedef __attribute__((ext_vector_type(8)))  float  v8f;

#define H 256
#define G 2048
#define NTASK 3
#define BN_EPS 1e-5f

__device__ __forceinline__ unsigned short f2bf_bits(float f) {
    unsigned u = __float_as_uint(f);
    unsigned r = u + 0x7FFFu + ((u >> 16) & 1u);   // round-to-nearest-even
    return (unsigned short)(r >> 16);
}

// ---------------- degree / normalization -----------------------------------
__global__ void degree_kernel(const int* __restrict__ dst, float* __restrict__ deg, int E) {
    int e = blockIdx.x * blockDim.x + threadIdx.x;
    if (e < E) atomicAdd(&deg[dst[e]], 1.0f);
}

__global__ void isqrt_kernel(const float* __restrict__ deg, float* __restrict__ dis, int N) {
    int i = blockIdx.x * blockDim.x + threadIdx.x;
    if (i < N) dis[i] = rsqrtf(deg[i] + 1.0f);
}

// ---------------- layer-1 linear (K=8, scalar) ------------------------------
__global__ void lin1_kernel(const float* __restrict__ x, const float* __restrict__ W1,
                            float* __restrict__ out, int N) {
    int idx = blockIdx.x * blockDim.x + threadIdx.x;
    if (idx >= N * H) return;
    int n = idx >> 8, f = idx & 255;
    const float* xr = x + n * 8;
    float acc = 0.f;
#pragma unroll
    for (int k = 0; k < 8; ++k) acc += xr[k] * W1[k * H + f];
    out[idx] = acc;
}

// ---------------- weight repack into WMMA B-fragment layout -----------------
// Fragment index = ((kt*16 + nt)*32 + lane)*16 + e
// lane<16 : col = nt*16+lane, K = kt*32 + e        (K 0..15)
// lane>=16: col = nt*16+lane-16, K = kt*32 + 16+e  (K 16..31)
__global__ void packw_kernel(const float* __restrict__ W, unsigned short* __restrict__ Wp) {
    int idx = blockIdx.x * blockDim.x + threadIdx.x;   // 8*16*32*16 = 65536
    if (idx >= 65536) return;
    int e    = idx & 15;
    int lane = (idx >> 4) & 31;
    int nt   = (idx >> 9) & 15;
    int kt   = idx >> 13;
    int col  = nt * 16 + (lane & 15);
    int K    = kt * 32 + (lane >> 4) * 16 + e;
    Wp[idx]  = f2bf_bits(W[(size_t)K * H + col]);
}

// ---------------- WMMA GEMM: C[NP,256] = A_bf16[NP,256] @ W_bf16[256,256] ---
// One wave: 16 rows x 128 cols (8 accum tiles = 64 VGPRs, no spills).
// 8 waves/block: 4 row-tiles x 2 column-halves -> 64 rows x 256 cols.
__global__ __launch_bounds__(256) void gemm256_wmma(const unsigned short* __restrict__ A,
                                                    const unsigned short* __restrict__ Wp,
                                                    float* __restrict__ C) {
    const int lane  = threadIdx.x;           // 0..31
    const int wave  = threadIdx.y;           // 0..7
    const int rtile = wave >> 1;             // 0..3
    const int chalf = wave & 1;              // 0..1 (columns 0-127 / 128-255)
    const int rowBase = blockIdx.x * 64 + rtile * 16;
    const int half = lane >> 4;              // 0 or 1
    const int l15  = lane & 15;

    v8f acc[8];
    v8f zero = {};
#pragma unroll
    for (int t = 0; t < 8; ++t) acc[t] = zero;

    const unsigned short* aRow = A + (size_t)(rowBase + l15) * H;

    for (int kt = 0; kt < 8; ++kt) {
        // A fragment: lane<16 -> K {kb..kb+15} with kb=kt*32
        //             lane>=16 -> shifted by 16 (half*16)
        union { v16bf v; uint4 u[2]; } a;
        const int kb = kt * 32 + half * 16;
        a.u[0] = *(const uint4*)(aRow + kb);
        a.u[1] = *(const uint4*)(aRow + kb + 8);

        const unsigned short* wTile =
            Wp + (size_t)(kt * 16 + chalf * 8) * 512 + lane * 16;
#pragma unroll
        for (int nt = 0; nt < 8; ++nt) {
            union { v16bf v; uint4 u[2]; } b;
            const unsigned short* wp = wTile + nt * 512;
            b.u[0] = *(const uint4*)(wp);
            b.u[1] = *(const uint4*)(wp + 8);
            acc[nt] = __builtin_amdgcn_wmma_f32_16x16x32_bf16(
                false, a.v, false, b.v, (short)0, acc[nt], false, false);
        }
    }

    // C/D layout: VGPR r, lanes0-15 -> M=r, lanes16-31 -> M=r+8; N=lane&15
#pragma unroll
    for (int nt = 0; nt < 8; ++nt) {
        float* cp = C + (size_t)(rowBase + half * 8) * H + chalf * 128 + nt * 16 + l15;
#pragma unroll
        for (int r = 0; r < 8; ++r) cp[(size_t)r * H] = acc[nt][r];
    }
}

// ---------------- edge scatter: agg[dst] += h[src]*norm ---------------------
__global__ void scatter_kernel(const int* __restrict__ src, const int* __restrict__ dst,
                               const float* __restrict__ dis, const float* __restrict__ h,
                               float* __restrict__ agg, int E) {
    int idx = blockIdx.x * blockDim.x + threadIdx.x;   // E * 64 threads
    if (idx >= E * 64) return;
    int e = idx >> 6;
    int c = (idx & 63) * 4;
    int s = src[e], d = dst[e];
    float nrm = dis[s] * dis[d];
    const float4 v = *(const float4*)(h + (size_t)s * H + c);
    float* o = agg + (size_t)d * H + c;
    atomicAdd(o + 0, v.x * nrm);
    atomicAdd(o + 1, v.y * nrm);
    atomicAdd(o + 2, v.z * nrm);
    atomicAdd(o + 3, v.w * nrm);
}

// ---------------- self-loop + bias ------------------------------------------
__global__ void finalize_kernel(const float* __restrict__ h, const float* __restrict__ dis,
                                const float* __restrict__ bias, float* __restrict__ agg, int N) {
    int idx = blockIdx.x * blockDim.x + threadIdx.x;
    if (idx >= N * H) return;
    int n = idx >> 8, f = idx & 255;
    float dii = dis[n] * dis[n];
    agg[idx] += h[idx] * dii + bias[f];
}

// ---------------- BN column stats (sum, sumsq) ------------------------------
__global__ void bn_stats_kernel(const float* __restrict__ v, float* __restrict__ sums, int N) {
    int col = threadIdx.x;                 // 256 threads = 256 columns
    int r0 = blockIdx.x * 128;
    int rend = r0 + 128; if (rend > N) rend = N;
    float s = 0.f, ss = 0.f;
    for (int r = r0; r < rend; ++r) {
        float x = v[(size_t)r * H + col];
        s += x; ss += x * x;
    }
    atomicAdd(&sums[col], s);
    atomicAdd(&sums[H + col], ss);
}

// ---------------- BN apply + ReLU + bf16 cast (zero pad rows) ---------------
__global__ void bn_apply_kernel(float* __restrict__ v, const float* __restrict__ sums,
                                const float* __restrict__ gamma, const float* __restrict__ beta,
                                unsigned short* __restrict__ hb, int N) {
    int n = blockIdx.x;                    // grid = NP blocks
    int f = threadIdx.x;                   // 256 threads
    size_t idx = (size_t)n * H + f;
    float y = 0.f;
    if (n < N) {
        float invN = 1.0f / (float)N;
        float mean = sums[f] * invN;
        float var  = sums[H + f] * invN - mean * mean;
        y = gamma[f] * (v[idx] - mean) * rsqrtf(var + BN_EPS) + beta[f];
        y = y > 0.f ? y : 0.f;
    }
    v[idx]  = y;
    hb[idx] = f2bf_bits(y);
}

// ---------------- pooling ----------------------------------------------------
__global__ void cnt_kernel(const int* __restrict__ batch, float* __restrict__ cnt, int N) {
    int i = blockIdx.x * blockDim.x + threadIdx.x;
    if (i < N) atomicAdd(&cnt[batch[i]], 1.0f);
}

__global__ void pool_kernel(const float* __restrict__ h, const int* __restrict__ batch,
                            float* __restrict__ psum, float* __restrict__ pmax, int N) {
    int idx = blockIdx.x * blockDim.x + threadIdx.x;
    if (idx >= N * H) return;
    int n = idx >> 8, f = idx & 255;
    int g = batch[n];
    float v = h[idx];                                  // post-ReLU, v >= 0
    atomicAdd(&psum[(size_t)g * H + f], v);
    atomicMax((int*)&pmax[(size_t)g * H + f], __float_as_int(v));
}

// ---------------- graph MLPs -------------------------------------------------
__global__ void s1_kernel(const float* __restrict__ psum, const float* __restrict__ pmax,
                          const float* __restrict__ cnt, const float* __restrict__ sW1,
                          const float* __restrict__ sb1, float* __restrict__ s1) {
    int idx = blockIdx.x * blockDim.x + threadIdx.x;   // G*256
    if (idx >= G * 256) return;
    int g = idx >> 8, o = idx & 255;
    float c = cnt[g];
    float inv = 1.0f / (c > 1.0f ? c : 1.0f);
    float acc = sb1[o];
    const float* ps = psum + (size_t)g * H;
    const float* pm = pmax + (size_t)g * H;
    for (int f = 0; f < H; ++f) {
        acc += ps[f] * inv * sW1[(size_t)f * 256 + o];
        acc += pm[f]       * sW1[(size_t)(H + f) * 256 + o];
    }
    s1[idx] = acc > 0.f ? acc : 0.f;
}

__global__ void s2_kernel(const float* __restrict__ s1, const float* __restrict__ sW2,
                          const float* __restrict__ sb2, float* __restrict__ s2) {
    int idx = blockIdx.x * blockDim.x + threadIdx.x;   // G*128
    if (idx >= G * 128) return;
    int g = idx >> 7, o = idx & 127;
    float acc = sb2[o];
    const float* sr = s1 + (size_t)g * 256;
    for (int f = 0; f < 256; ++f) acc += sr[f] * sW2[(size_t)f * 128 + o];
    s2[idx] = acc > 0.f ? acc : 0.f;
}

__global__ void head_kernel(const float* __restrict__ s2, const float* __restrict__ hW1,
                            const float* __restrict__ hb1, const float* __restrict__ hW2,
                            const float* __restrict__ hb2, float* __restrict__ out) {
    int idx = blockIdx.x * blockDim.x + threadIdx.x;   // NTASK*G
    if (idx >= NTASK * G) return;
    int t = idx / G, g = idx % G;
    const float* sr = s2 + (size_t)g * 128;
    float res = hb2[t];
    for (int o = 0; o < 32; ++o) {
        float a = hb1[t * 32 + o];
        const float* w = hW1 + (size_t)t * 128 * 32 + o;
        for (int f = 0; f < 128; ++f) a += sr[f] * w[(size_t)f * 32];
        a = a > 0.f ? a : 0.f;
        res += a * hW2[t * 32 + o];
    }
    out[idx] = res;                                    // [task, graph]
}

// ---------------------------------------------------------------------------
extern "C" void kernel_launch(void* const* d_in, const int* in_sizes, int n_in,
                              void* d_out, int out_size, void* d_ws, size_t ws_size,
                              hipStream_t stream) {
    const float* x     = (const float*)d_in[0];
    const int*   ei    = (const int*)d_in[1];
    const int*   batch = (const int*)d_in[2];
    const float* W1  = (const float*)d_in[3];
    const float* b1  = (const float*)d_in[4];
    const float* Wl[3] = { (const float*)d_in[5], (const float*)d_in[7], (const float*)d_in[9] };
    const float* bl[3] = { (const float*)d_in[6], (const float*)d_in[8], (const float*)d_in[10] };
    const float* gm[4] = { (const float*)d_in[11], (const float*)d_in[13],
                           (const float*)d_in[15], (const float*)d_in[17] };
    const float* be[4] = { (const float*)d_in[12], (const float*)d_in[14],
                           (const float*)d_in[16], (const float*)d_in[18] };
    const float* sW1 = (const float*)d_in[19];
    const float* sb1 = (const float*)d_in[20];
    const float* sW2 = (const float*)d_in[21];
    const float* sb2 = (const float*)d_in[22];
    const float* hW1 = (const float*)d_in[23];
    const float* hb1 = (const float*)d_in[24];
    const float* hW2 = (const float*)d_in[25];
    const float* hb2 = (const float*)d_in[26];
    float* out = (float*)d_out;

    const int N  = in_sizes[2];
    const int E  = in_sizes[1] / 2;
    const int NP = (N + 127) & ~127;       // pad rows to 128 (also multiple of 64)

    // ---- workspace carve ----
    char* p = (char*)d_ws;
    auto carve = [&](size_t bytes) { void* r = (void*)p; p += (bytes + 255) & ~(size_t)255; return r; };
    float*          bufA   = (float*)carve((size_t)NP * H * 4);     // linear output
    float*          bufB   = (float*)carve((size_t)NP * H * 4);     // aggregate / BN output
    unsigned short* bufH   = (unsigned short*)carve((size_t)NP * H * 2); // bf16 activations
    float*          deg    = (float*)carve((size_t)N * 4);
    float*          dis    = (float*)carve((size_t)N * 4);
    float*          sums   = (float*)carve(2 * H * 4);
    unsigned short* Wp     = (unsigned short*)carve((size_t)3 * H * H * 2);
    float*          psum   = (float*)carve((size_t)G * H * 4);
    float*          pmax   = (float*)carve((size_t)G * H * 4);
    float*          cnt    = (float*)carve((size_t)G * 4);
    float*          s1     = (float*)carve((size_t)G * 256 * 4);
    float*          s2     = (float*)carve((size_t)G * 128 * 4);
    (void)ws_size; (void)n_in; (void)out_size;

    const int* src = ei;
    const int* dst = ei + E;

    // ---- degree / D^{-1/2} ----
    hipMemsetAsync(deg, 0, (size_t)N * 4, stream);
    degree_kernel<<<(E + 255) / 256, 256, 0, stream>>>(dst, deg, E);
    isqrt_kernel<<<(N + 255) / 256, 256, 0, stream>>>(deg, dis, N);

    // ---- pack W2..W4 into WMMA B-fragment layout ----
    for (int l = 0; l < 3; ++l)
        packw_kernel<<<65536 / 256, 256, 0, stream>>>(Wl[l], Wp + (size_t)l * H * H);

    // ---- per-layer post-linear pipeline ----
    auto post = [&](const float* bias, const float* gamma, const float* beta) {
        hipMemsetAsync(bufB, 0, (size_t)NP * H * 4, stream);
        scatter_kernel<<<(E * 64 + 255) / 256, 256, 0, stream>>>(src, dst, dis, bufA, bufB, E);
        finalize_kernel<<<(N * H + 255) / 256, 256, 0, stream>>>(bufA, dis, bias, bufB, N);
        hipMemsetAsync(sums, 0, 2 * H * 4, stream);
        bn_stats_kernel<<<(N + 127) / 128, 256, 0, stream>>>(bufB, sums, N);
        bn_apply_kernel<<<NP, 256, 0, stream>>>(bufB, sums, gamma, beta, bufH, N);
    };

    // layer 1 (K=8 scalar linear)
    lin1_kernel<<<(N * H + 255) / 256, 256, 0, stream>>>(x, W1, bufA, N);
    post(b1, gm[0], be[0]);

    // layers 2..4 (WMMA bf16 GEMMs)
    for (int l = 0; l < 3; ++l) {
        gemm256_wmma<<<NP / 64, dim3(32, 8), 0, stream>>>(bufH, Wp + (size_t)l * H * H, bufA);
        post(bl[l], gm[l + 1], be[l + 1]);
    }

    // ---- pooling (bufB holds final fp32 node features) ----
    hipMemsetAsync(psum, 0, (size_t)G * H * 4, stream);
    hipMemsetAsync(pmax, 0, (size_t)G * H * 4, stream);
    hipMemsetAsync(cnt,  0, (size_t)G * 4, stream);
    cnt_kernel<<<(N + 255) / 256, 256, 0, stream>>>(batch, cnt, N);
    pool_kernel<<<(N * H + 255) / 256, 256, 0, stream>>>(bufB, batch, psum, pmax, N);

    // ---- graph MLPs + heads ----
    s1_kernel<<<(G * 256 + 255) / 256, 256, 0, stream>>>(psum, pmax, cnt, sW1, sb1, s1);
    s2_kernel<<<(G * 128 + 255) / 256, 256, 0, stream>>>(s1, sW2, sb2, s2);
    head_kernel<<<(NTASK * G + 255) / 256, 256, 0, stream>>>(s2, hW1, hb1, hW2, hb2, out);
}
